// BertAttention_26156350832687
// MI455X (gfx1250) — matrix operand order repaired and checked by reading, not compiled, not run on hardware
//
#include <hip/hip_runtime.h>
#include <hip/hip_bf16.h>
#include <stdint.h>

// ---------- types ----------
typedef __attribute__((ext_vector_type(16))) __bf16          v16bf;
typedef __attribute__((ext_vector_type(8)))  float           v8f;
typedef __attribute__((ext_vector_type(8)))  unsigned short  us8;   // 16 B
typedef __attribute__((ext_vector_type(16))) unsigned short  us16;  // 32 B
typedef __attribute__((ext_vector_type(4)))  unsigned short  us4;

union ABFrag {
    us16  u;
    v16bf b;
    us8   h[2];
    unsigned short s[16];
};

// fp32 -> bf16 round-to-nearest-even
__device__ __forceinline__ unsigned short f2bf(float f) {
    unsigned int u = __float_as_uint(f);
    unsigned int r = u + 0x7FFFu + ((u >> 16) & 1u);
    return (unsigned short)(r >> 16);
}

// LDS byte offset of a __shared__ pointer (low 32 bits of the generic addr).
__device__ __forceinline__ unsigned lds_off(const void* p) {
    return (unsigned)(uintptr_t)p;
}

// gfx1250 async global->LDS copy, 16B per lane, tracked by ASYNCcnt
#define ASYNC_G2L_B128(ldsoff, gptr)                                          \
    asm volatile("global_load_async_to_lds_b128 %0, %1, off"                  \
                 :: "v"(ldsoff), "v"(gptr) : "memory")
#define WAIT_ASYNC(n) asm volatile("s_wait_asynccnt " #n ::: "memory")

// single-instruction xor-butterfly step: ds_swizzle_b32 group-of-32 mode,
// lane' = ((lane & 0x1f) | 0) ^ m   (m in {1,2,4,8} stays inside 16-lane groups)
#define SWIZ_XOR(x, m) \
    __int_as_float(__builtin_amdgcn_ds_swizzle(__float_as_int(x), (((m) << 10) | 0x1f)))

__device__ __forceinline__ float rowmax16(float x) {
    x = fmaxf(x, SWIZ_XOR(x, 1));
    x = fmaxf(x, SWIZ_XOR(x, 2));
    x = fmaxf(x, SWIZ_XOR(x, 4));
    x = fmaxf(x, SWIZ_XOR(x, 8));
    return x;
}
__device__ __forceinline__ float rowsum16(float x) {
    x += SWIZ_XOR(x, 1);
    x += SWIZ_XOR(x, 2);
    x += SWIZ_XOR(x, 4);
    x += SWIZ_XOR(x, 8);
    return x;
}

// ---------- problem constants ----------
#define BB    4
#define SS    2048
#define DD    1024
#define HH    16
#define HEADD 64
#define SCALE (1.0f / 32.0f)   // 1/sqrt(1024)

// ---------- fp32 -> bf16 conversion ----------
__global__ void cvt_bf16_kernel(const float* __restrict__ in,
                                unsigned short* __restrict__ out, int n) {
    int i = (blockIdx.x * blockDim.x + threadIdx.x) * 4;
    if (i + 3 < n) {
        float4 f = *(const float4*)(in + i);
        us4 o;
        o.x = f2bf(f.x); o.y = f2bf(f.y); o.z = f2bf(f.z); o.w = f2bf(f.w);
        *(us4*)(out + i) = o;
    }
}

// ---------- Y[M,N] = X[M,K] @ W[N,K]^T  (bf16 in, fp32 acc, bf16/fp32 out) ----
// Block: 256 threads = 8 waves; macro tile 128(M) x 128(N); wave owns 32x64.
// K consumed in 32-wide slices, double-buffered in LDS via async DMA.
template <bool F32OUT>
__global__ void gemm_bf16_kernel(const unsigned short* __restrict__ X,
                                 const unsigned short* __restrict__ W,
                                 void* __restrict__ Y,
                                 int M, int N, int K) {
    __shared__ __align__(16) unsigned short Asm[2][128 * 32];  // 2 x 8 KB
    __shared__ __align__(16) unsigned short Bsm[2][128 * 32];  // 2 x 8 KB

    const int tid  = threadIdx.x;
    const int wid  = tid >> 5;
    const int lane = tid & 31;
    const int g    = lane >> 4;
    const int ln   = lane & 15;

    const int m0 = blockIdx.y * 128;
    const int n0 = blockIdx.x * 128;
    const int wm = (wid & 3) * 32;      // wave M offset (4 waves along M)
    const int wn = (wid >> 2) * 64;     // wave N offset (2 waves along N)

    v8f c[2][4];
    for (int i = 0; i < 2; ++i)
        for (int j = 0; j < 4; ++j)
            c[i][j] = (v8f){};

    const int nsteps = K / 32;  // 32

    // per-thread DMA (tile row = 32 halves = 64 B):
    //   A tile 128x32 halves = 8 KB -> 2 x b128/thread
    //   B tile 128x32 halves = 8 KB -> 2 x b128/thread
    auto issue_stage = [&](int kk, int buf) {
        for (int cchunk = 0; cchunk < 2; ++cchunk) {
            int byteoff = (tid + cchunk * 256) * 16;
            int row  = byteoff >> 6;        // 0..127
            int colb = byteoff & 63;
            const char* ga = (const char*)(X + (size_t)(m0 + row) * K + kk) + colb;
            const char* gb = (const char*)(W + (size_t)(n0 + row) * K + kk) + colb;
            ASYNC_G2L_B128(lds_off(&Asm[buf][byteoff >> 1]), ga);
            ASYNC_G2L_B128(lds_off(&Bsm[buf][byteoff >> 1]), gb);
        }
    };

    issue_stage(0, 0);

    for (int s = 0; s < nsteps; ++s) {
        const int buf = s & 1;
        if (s + 1 < nsteps) {
            issue_stage((s + 1) * 32, (s + 1) & 1);
            WAIT_ASYNC(4);   // 4 next-stage DMAs in flight; stage s complete
        } else {
            WAIT_ASYNC(0);
        }
        __syncthreads();     // stage s visible to all waves

        ABFrag a[2], b[4];
        for (int i = 0; i < 2; ++i) {
            const unsigned short* p = &Asm[buf][(wm + i * 16 + ln) * 32 + g * 8];
            a[i].h[0] = *(const us8*)(p);
            a[i].h[1] = *(const us8*)(p + 16);
        }
        for (int j = 0; j < 4; ++j) {
            const unsigned short* q = &Bsm[buf][(wn + j * 16 + ln) * 32 + g * 8];
            b[j].h[0] = *(const us8*)(q);
            b[j].h[1] = *(const us8*)(q + 16);
        }

        for (int i = 0; i < 2; ++i)
            for (int j = 0; j < 4; ++j)
                c[i][j] = __builtin_amdgcn_wmma_f32_16x16x32_bf16(
                    false, a[i].b, false, b[j].b, (short)0, c[i][j], false, false);

        __syncthreads();     // all reads of buf done before stage s+2 overwrites
    }

    // C layout: row = vgpr + 8*g, col = ln
    for (int ti = 0; ti < 2; ++ti)
        for (int tj = 0; tj < 4; ++tj)
            for (int v = 0; v < 8; ++v) {
                int m = m0 + wm + ti * 16 + v + 8 * g;
                int n = n0 + wn + tj * 16 + ln;
                if (F32OUT)
                    ((float*)Y)[(size_t)m * N + n] = c[ti][tj][v];
                else
                    ((unsigned short*)Y)[(size_t)m * N + n] = f2bf(c[ti][tj][v]);
            }
}

// ---------- flash attention ----------
// Block: 256 threads = 8 waves; all 8 waves share one (b,h) and walk the same
// 32-key tiles, so K/V tiles are staged once per block in LDS (async, double
// buffered). Each wave owns a 16-query tile.
__global__ void attn_kernel(const unsigned short* __restrict__ Qm,
                            const unsigned short* __restrict__ Km,
                            const unsigned short* __restrict__ Vm,
                            unsigned short* __restrict__ Ctx) {
    __shared__ __align__(16) unsigned short Kt[2][32 * 64];   // 2 x 4 KB
    __shared__ __align__(16) unsigned short Vt[2][32 * 64];   // 2 x 4 KB
    __shared__ __align__(16) unsigned short pbuf[8][16 * 32]; // 8 KB

    const int tid  = threadIdx.x;
    const int wid  = tid >> 5;
    const int lane = tid & 31;
    const int g    = lane >> 4;
    const int ln   = lane & 15;

    const int gw    = blockIdx.x * 8 + wid;   // 0..8191
    const int qtile = gw & 127;
    const int bh    = gw >> 7;                // identical for all 8 waves
    const int b     = bh >> 4;
    const int h     = bh & 15;

    const size_t bhoff = (size_t)b * SS * DD + (size_t)h * HEADD;
    const unsigned short* Qb = Qm + bhoff;
    const unsigned short* Kb = Km + bhoff;
    const unsigned short* Vb = Vm + bhoff;

    const int qrow0 = qtile * 16;

    // Q A-fragments for k = 0..31 and 32..63 (lane = query row)
    ABFrag aq[2];
    for (int ks = 0; ks < 2; ++ks) {
        const unsigned short* p = Qb + (size_t)(qrow0 + ln) * DD + ks * 32 + g * 8;
        aq[ks].h[0] = *(const us8*)(p);
        aq[ks].h[1] = *(const us8*)(p + 16);
    }

    v8f o[4];
    o[0] = (v8f){}; o[1] = (v8f){}; o[2] = (v8f){}; o[3] = (v8f){};
    float m_i[8], l_i[8];
    for (int v = 0; v < 8; ++v) { m_i[v] = -INFINITY; l_i[v] = 0.0f; }

    unsigned short* pb = pbuf[wid];

    // per-thread DMA: K tile 32x64 halves (row = 128 B) -> 1 b128/thread; same for V
    auto issue_tiles = [&](int tbase, int buf) {
        int byteoff = tid * 16;
        int row  = byteoff >> 7;    // 0..31
        int colb = byteoff & 127;
        const char* gk = (const char*)(Kb + (size_t)(tbase + row) * DD) + colb;
        const char* gv = (const char*)(Vb + (size_t)(tbase + row) * DD) + colb;
        ASYNC_G2L_B128(lds_off(&Kt[buf][byteoff >> 1]), gk);
        ASYNC_G2L_B128(lds_off(&Vt[buf][byteoff >> 1]), gv);
    };

    const int niter = SS / 32;  // 64
    issue_tiles(0, 0);

    for (int it = 0; it < niter; ++it) {
        const int buf = it & 1;
        if (it + 1 < niter) {
            issue_tiles((it + 1) * 32, (it + 1) & 1);
            WAIT_ASYNC(2);
        } else {
            WAIT_ASYNC(0);
        }
        __syncthreads();   // K/V tiles for stage `it` visible

        // ---- scores from LDS K tile ----
        v8f s0 = (v8f){}, s1 = (v8f){};
        for (int ks = 0; ks < 2; ++ks) {
            ABFrag bk0, bk1;
            const int kb = ks * 32 + g * 8;
            bk0.h[0] = *(const us8*)&Kt[buf][(ln)      * 64 + kb];
            bk0.h[1] = *(const us8*)&Kt[buf][(ln)      * 64 + kb + 16];
            bk1.h[0] = *(const us8*)&Kt[buf][(16 + ln) * 64 + kb];
            bk1.h[1] = *(const us8*)&Kt[buf][(16 + ln) * 64 + kb + 16];
            s0 = __builtin_amdgcn_wmma_f32_16x16x32_bf16(false, aq[ks].b, false, bk0.b, (short)0, s0, false, false);
            s1 = __builtin_amdgcn_wmma_f32_16x16x32_bf16(false, aq[ks].b, false, bk1.b, (short)0, s1, false, false);
        }

        // ---- online softmax (ds_swizzle butterflies within 16-lane groups) ----
        for (int v = 0; v < 8; ++v) {
            float a   = s0[v] * SCALE;
            float bsc = s1[v] * SCALE;
            float mx   = rowmax16(fmaxf(a, bsc));
            float newm = fmaxf(m_i[v], mx);
            float corr = __expf(m_i[v] - newm);
            float p0 = __expf(a   - newm);
            float p1 = __expf(bsc - newm);
            float rs = rowsum16(p0 + p1);
            l_i[v] = l_i[v] * corr + rs;
            m_i[v] = newm;
            o[0][v] *= corr; o[1][v] *= corr; o[2][v] *= corr; o[3][v] *= corr;
            s0[v] = p0; s1[v] = p1;
        }

        // ---- stage P (16x32 bf16) in LDS: C-layout -> row-major ----
        for (int v = 0; v < 8; ++v) {
            int r = v + 8 * g;
            pb[r * 32 + ln]      = f2bf(s0[v]);
            pb[r * 32 + 16 + ln] = f2bf(s1[v]);
        }
        __syncthreads();   // P visible / ordered

        // ---- P as A-fragment, V B-fragments gathered from LDS V tile ----
        ABFrag ap;
        ap.h[0] = *(const us8*)&pb[ln * 32 + g * 8];
        ap.h[1] = *(const us8*)&pb[ln * 32 + 16 + g * 8];

        for (int dt = 0; dt < 4; ++dt) {
            ABFrag bv;
            const int d = dt * 16 + ln;
            for (int hh = 0; hh < 16; ++hh)
                bv.s[hh] = Vt[buf][((hh < 8) ? (g * 8 + hh) : (16 + g * 8 + hh - 8)) * 64 + d];
            o[dt] = __builtin_amdgcn_wmma_f32_16x16x32_bf16(false, ap.b, false, bv.b, (short)0, o[dt], false, false);
        }
        __syncthreads();   // done reading buf before stage it+2 overwrites it
    }

    // ---- normalize + store ctx ([B,S,D] merged-head layout) ----
    for (int dt = 0; dt < 4; ++dt)
        for (int v = 0; v < 8; ++v) {
            int r = v + 8 * g;
            float val = o[dt][v] * (1.0f / l_i[v]);
            Ctx[bhoff + (size_t)(qrow0 + r) * DD + dt * 16 + ln] = f2bf(val);
        }
}

// ---------- launcher ----------
extern "C" void kernel_launch(void* const* d_in, const int* in_sizes, int n_in,
                              void* d_out, int out_size, void* d_ws, size_t ws_size,
                              hipStream_t stream) {
    const float* hidden = (const float*)d_in[0];
    // d_in[1] = attention_mask (unused, faithful to reference)
    const float* w_q = (const float*)d_in[2];
    const float* w_k = (const float*)d_in[3];
    const float* w_v = (const float*)d_in[4];
    const float* w_o = (const float*)d_in[5];

    const int M = BB * SS;          // 8192
    const int N = DD;               // 1024
    const int K = DD;               // 1024
    const int nHidden = M * DD;     // 8,388,608
    const int nW      = DD * DD;    // 1,048,576

    // workspace layout (bf16 buffers), total 88 MB
    unsigned short* ws = (unsigned short*)d_ws;
    unsigned short* Xb  = ws;                                  // 16 MB
    unsigned short* Wq  = Xb  + (size_t)nHidden;               //  2 MB each
    unsigned short* Wk  = Wq  + (size_t)nW;
    unsigned short* Wv  = Wk  + (size_t)nW;
    unsigned short* Wo  = Wv  + (size_t)nW;
    unsigned short* Qb  = Wo  + (size_t)nW;                    // 16 MB each
    unsigned short* Kb  = Qb  + (size_t)nHidden;
    unsigned short* Vb  = Kb  + (size_t)nHidden;
    unsigned short* Ctx = Vb  + (size_t)nHidden;

    // 1) convert fp32 -> bf16
    {
        dim3 blk(256);
        cvt_bf16_kernel<<<dim3((nHidden / 4 + 255) / 256), blk, 0, stream>>>(hidden, Xb, nHidden);
        cvt_bf16_kernel<<<dim3((nW / 4 + 255) / 256), blk, 0, stream>>>(w_q, Wq, nW);
        cvt_bf16_kernel<<<dim3((nW / 4 + 255) / 256), blk, 0, stream>>>(w_k, Wk, nW);
        cvt_bf16_kernel<<<dim3((nW / 4 + 255) / 256), blk, 0, stream>>>(w_v, Wv, nW);
        cvt_bf16_kernel<<<dim3((nW / 4 + 255) / 256), blk, 0, stream>>>(w_o, Wo, nW);
    }

    // 2) Q/K/V projections (y = x @ W^T)
    dim3 ggrid(N / 128, M / 128);  // (8, 64)
    gemm_bf16_kernel<false><<<ggrid, 256, 0, stream>>>(Xb, Wq, Qb, M, N, K);
    gemm_bf16_kernel<false><<<ggrid, 256, 0, stream>>>(Xb, Wk, Kb, M, N, K);
    gemm_bf16_kernel<false><<<ggrid, 256, 0, stream>>>(Xb, Wv, Vb, M, N, K);

    // 3) attention (flash-style; K/V tiles async-staged in LDS per block)
    attn_kernel<<<dim3(1024), 256, 0, stream>>>(Qb, Kb, Vb, Ctx);

    // 4) output projection, fp32 straight into d_out
    gemm_bf16_kernel<true><<<ggrid, 256, 0, stream>>>(Ctx, Wo, d_out, M, N, K);
}
